// S2Block_13821204758832
// MI455X (gfx1250) — matrix operand order repaired and checked by reading, not compiled, not run on hardware
//
#include <hip/hip_runtime.h>

// ---------------- problem constants ----------------
constexpr int kB     = 2;
constexpr int kC     = 64;       // DIM
constexpr int kN     = 2304;     // 48*48 tokens
constexpr int kHeads = 4;
constexpr int kMLP   = 256;
constexpr int kRows  = kB * kN;  // 4608
constexpr float kLog2e   = 1.4426950408889634f;
constexpr float kScale   = 0.25f;                    // DIM_HEAD^-0.5
constexpr float kScaleSE = 0.25f / 144.0f;           // scale / (n / DIM_HEAD)

typedef __attribute__((ext_vector_type(16))) _Float16 v16h;
typedef __attribute__((ext_vector_type(8)))  _Float16 v8h;
typedef __attribute__((ext_vector_type(8)))  float    v8f;

__device__ __forceinline__ float fast_exp2(float x) {
  return __builtin_amdgcn_exp2f(x);   // bare v_exp_f32 (exp2), no libm fixup
}

__device__ __forceinline__ v8f zero8() {
  v8f z = {0.f,0.f,0.f,0.f,0.f,0.f,0.f,0.f};
  return z;
}
__device__ __forceinline__ v8h zero8h() {
  v8h z = {(_Float16)0.f,(_Float16)0.f,(_Float16)0.f,(_Float16)0.f,
           (_Float16)0.f,(_Float16)0.f,(_Float16)0.f,(_Float16)0.f};
  return z;
}
__device__ __forceinline__ v16h zero16h() {
  v8h z = zero8h();
  return __builtin_shufflevector(z, z, 0,1,2,3,4,5,6,7,0,1,2,3,4,5,6,7);
}

__device__ __forceinline__ v8f wmma_f16(v16h a, v16h b, v8f c) {
  return __builtin_amdgcn_wmma_f32_16x16x32_f16(false, a, false, b, (short)0, c,
                                                false, false);
}

// A fragment (16x32 f16) from row-major source, stride ld, origin (row0,k0).
__device__ __forceinline__ v16h load_A(const _Float16* __restrict__ src,
                                       int row0, int k0, int ld) {
  int lane = threadIdx.x & 31;
  int m  = lane & 15;
  int kb = (lane >> 4) << 3;
  const _Float16* p = src + (size_t)(row0 + m) * ld + k0 + kb;
  v8h lo = *(const v8h*)p;
  v8h hi = *(const v8h*)(p + 16);
  return __builtin_shufflevector(lo, hi, 0,1,2,3,4,5,6,7,8,9,10,11,12,13,14,15);
}

// A fragment with K truncated to 16 (upper 16 K zero) from row-major f16 source.
__device__ __forceinline__ v16h load_A_pad16(const _Float16* __restrict__ src,
                                             int row0, int col0, int ld) {
  int lane = threadIdx.x & 31;
  int m  = lane & 15;
  int kb = (lane >> 4) << 3;
  const _Float16* p = src + (size_t)(row0 + m) * ld + col0 + kb;
  v8h lo = *(const v8h*)p;
  v8h z  = zero8h();
  return __builtin_shufflevector(lo, z, 0,1,2,3,4,5,6,7,8,9,10,11,12,13,14,15);
}

// B fragment from pre-swizzled matrix: one contiguous 32B per-lane load.
// Layout: [((kt*ntiles + ct)*32 + lane)*16 + h]
__device__ __forceinline__ v16h load_B_sw(const _Float16* __restrict__ swz,
                                          int kt, int ct, int ntiles) {
  int lane = threadIdx.x & 31;
  return *(const v16h*)(swz + ((size_t)((kt * ntiles + ct) * 32 + lane) << 4));
}

// butterfly reductions across the 16 lanes that hold one C-row (secm kernel)
__device__ __forceinline__ float rowmax16(float v) {
#pragma unroll
  for (int m = 1; m < 16; m <<= 1) v = fmaxf(v, __shfl_xor(v, m, 32));
  return v;
}
__device__ __forceinline__ float rowsum16(float v) {
#pragma unroll
  for (int m = 1; m < 16; m <<= 1) v += __shfl_xor(v, m, 32);
  return v;
}

// ---------------- K0: fp32 weight -> f16 B-fragment swizzle ----------------
__global__ void cvt_swz(const float* __restrict__ W, _Float16* __restrict__ dst,
                        int N, int ntiles, int nelem) {
  int tid = blockIdx.x * blockDim.x + threadIdx.x;
  if (tid >= nelem) return;
  int h    = tid & 15;
  int lane = (tid >> 4) & 31;
  int blk  = tid >> 9;
  int ct = blk % ntiles, kt = blk / ntiles;
  int k = kt * 32 + ((lane >> 4) << 3) + (h & 7) + ((h >> 3) << 4);
  dst[tid] = (_Float16)W[(size_t)k * N + ct * 16 + (lane & 15)];
}

// Swizzle x1 into per-head A^T-fragment layout for out1^T = X1^T @ P^T:
// A[m=d][k=j]; dst[(( (j/32)*4 + head )*32 + lane)*16 + h] with A k-pattern.
__global__ void swizzle_x1A(const _Float16* __restrict__ x1h,
                            _Float16* __restrict__ x1swA) {
  int tid = blockIdx.x * blockDim.x + threadIdx.x;  // < kRows*64
  int h    = tid & 15;
  int lane = (tid >> 4) & 31;
  int blk  = tid >> 9;
  int hh = blk & 3, kt = blk >> 2;
  int kA = (h & 7) + ((h >> 3) << 4) + ((lane >> 4) << 3);
  int row = kt * 32 + kA;                    // j (token index within 32-tile)
  x1swA[tid] = x1h[(size_t)row * 64 + hh * 16 + (lane & 15)];
}

// ---------------- K1: LayerNorm1 + transpose (vectorized writeback) ----------
__global__ void ln_kernel(const float* __restrict__ x, const float* __restrict__ y,
                          const float* __restrict__ g, const float* __restrict__ bb,
                          float* __restrict__ xt,
                          _Float16* __restrict__ xnh, _Float16* __restrict__ ynh) {
  int r = blockIdx.x * blockDim.x + threadIdx.x;
  if (r >= kRows) return;
  int b = r / kN, p = r % kN;
  const float* xb = x + (size_t)b * kC * kN + p;
  const float* yb = y + (size_t)b * kC * kN + p;
  float sx = 0.f, sx2 = 0.f, sy = 0.f, sy2 = 0.f;
#pragma unroll 1
  for (int c = 0; c < kC; ++c) {
    float vx = xb[(size_t)c * kN]; sx += vx; sx2 += vx * vx;
    float vy = yb[(size_t)c * kN]; sy += vy; sy2 += vy * vy;
  }
  float mx = sx * (1.f/64.f), vx = sx2 * (1.f/64.f) - mx * mx;
  float my = sy * (1.f/64.f), vy = sy2 * (1.f/64.f) - my * my;
  float rsx = rsqrtf(vx + 1e-5f), rsy = rsqrtf(vy + 1e-5f);
#pragma unroll 1
  for (int c0 = 0; c0 < kC; c0 += 8) {
    v8h xnv, ynv;
    float xtv[8];
#pragma unroll
    for (int i = 0; i < 8; ++i) {
      float vxx = xb[(size_t)(c0 + i) * kN];
      float vyy = yb[(size_t)(c0 + i) * kN];
      xtv[i] = vxx;
      xnv[i] = (_Float16)((vxx - mx) * rsx * g[c0 + i] + bb[c0 + i]);
      ynv[i] = (_Float16)((vyy - my) * rsy * g[c0 + i] + bb[c0 + i]);
    }
    float4 f0 = {xtv[0], xtv[1], xtv[2], xtv[3]};
    float4 f1 = {xtv[4], xtv[5], xtv[6], xtv[7]};
    *(float4*)(xt + (size_t)r * 64 + c0)     = f0;
    *(float4*)(xt + (size_t)r * 64 + c0 + 4) = f1;
    *(v8h*)(xnh + (size_t)r * 64 + c0) = xnv;
    *(v8h*)(ynh + (size_t)r * 64 + c0) = ynv;
  }
}

// ---------------- K2: projections, 16x64 strip per wave ----------------
// mat==1 (y2) is pre-scaled by kScale*log2(e): attention uses base-2 softmax.
__global__ void proj_kernel(const _Float16* __restrict__ xnh,
                            const _Float16* __restrict__ ynh,
                            const _Float16* __restrict__ wa1,
                            const _Float16* __restrict__ wa2,
                            const _Float16* __restrict__ we1,
                            const _Float16* __restrict__ we2,
                            _Float16* __restrict__ y1h, _Float16* __restrict__ y2h,
                            _Float16* __restrict__ x1h, _Float16* __restrict__ x2h) {
  int wgid = blockIdx.x * (blockDim.x >> 5) + (threadIdx.x >> 5);
  int mat = wgid & 3;
  int rt  = wgid >> 2;               // < 288
  const _Float16* src = (mat < 2) ? ynh : xnh;
  const _Float16* wgt = (mat == 0) ? wa1 : (mat == 1) ? wa2 : (mat == 2) ? we1 : we2;
  _Float16* dst = (mat == 0) ? y1h : (mat == 1) ? y2h : (mat == 2) ? x1h : x2h;
  float oscale = (mat == 1) ? kScale * kLog2e : 1.f;
  int row0 = rt * 16;
  v8f acc[4] = {zero8(), zero8(), zero8(), zero8()};
#pragma unroll
  for (int kt = 0; kt < 2; ++kt) {
    v16h a = load_A(src, row0, kt * 32, 64);
#pragma unroll
    for (int ct = 0; ct < 4; ++ct)
      acc[ct] = wmma_f16(a, load_B_sw(wgt, kt, ct, 4), acc[ct]);
  }
  int lane = threadIdx.x & 31, n = lane & 15, mb = (lane >> 4) << 3;
#pragma unroll
  for (int ct = 0; ct < 4; ++ct)
#pragma unroll
    for (int r = 0; r < 8; ++r)
      dst[(size_t)(row0 + mb + r) * 64 + ct * 16 + n] = (_Float16)(acc[ct][r] * oscale);
}

// ---------------- K2b: channel map, stored transposed ----------------
// secmh_t[j*16 + i] = softmax_j(x1^T x2 * s)[i][j]  (base-2 softmax)
__global__ void secm_kernel(const _Float16* __restrict__ x1h,
                            const _Float16* __restrict__ x2h,
                            _Float16* __restrict__ secmh_t) {
  int w = threadIdx.x >> 5;         // 8 waves, one per (b,h)
  int b = w >> 2, hh = w & 3;
  int lane = threadIdx.x & 31, n = lane & 15, m = lane & 15;
  int kbA = (lane >> 4) << 3, kbB = (lane >> 4) << 4, mb = (lane >> 4) << 3;
  size_t base = (size_t)b * kN * 64 + hh * 16;
  v8f c = zero8();
#pragma unroll 1
  for (int t = 0; t < kN / 32; ++t) {
    int k0 = t * 32;
    v16h a, bf;
#pragma unroll
    for (int i = 0; i < 8; ++i) {  // A[m][k] = x1[k][m] (transposed read)
      a[i]     = x1h[base + (size_t)(k0 + kbA + i)      * 64 + m];
      a[i + 8] = x1h[base + (size_t)(k0 + kbA + 16 + i) * 64 + m];
    }
#pragma unroll
    for (int h = 0; h < 16; ++h)
      bf[h] = x2h[base + (size_t)(k0 + kbA + (h & 7) + ((h >> 3) << 4)) * 64 + n];
    c = wmma_f16(a, bf, c);
  }
#pragma unroll
  for (int r = 0; r < 8; ++r) {    // softmax over j (across lanes in half-wave)
    float s  = c[r] * (kScaleSE * kLog2e);
    float mx = rowmax16(s);
    float e  = fast_exp2(s - mx);
    float sm = rowsum16(e);
    c[r] = e / sm;
  }
#pragma unroll
  for (int r = 0; r < 8; ++r)      // store transposed: [j*16 + i]
    secmh_t[(size_t)(b * kHeads + hh) * 256 + n * 16 + (mb + r)] = (_Float16)c[r];
}

// ---------------- K3: flash attention (fully transposed, wave-local) --------
// S^T = K_j @ Q^T ; out1^T = X1^T @ P^T ; out2^T = secm^T @ Q^T
// Per-token softmax stats are per-lane scalars; base-2 exp (scale folded in y2).
// P staging is wave-local: DS in-order per wave + explicit s_wait_dscnt, no
// block barrier -> the 4 waves per WGP run fully decoupled.
__global__ void attn_kernel(const _Float16* __restrict__ y1h,
                            const _Float16* __restrict__ y2h,
                            const _Float16* __restrict__ x1swA,
                            const _Float16* __restrict__ secmh_t,
                            _Float16* __restrict__ attnh) {
  __shared__ _Float16 Pl[4][16 * 32];   // P row-major [i][j] per wave
  int w = threadIdx.x >> 5;
  int wgid = blockIdx.x * 4 + w;
  int rt = wgid % 144;
  int hh = (wgid / 144) & 3;
  int b  = wgid / 576;
  int lane = threadIdx.x & 31, n = lane & 15;
  int kbB = (lane >> 4) << 4, mb = (lane >> 4) << 3;
  size_t rowBase = (size_t)b * kN + rt * 16;

  // Q^T B-fragment (K = head dim, zero-padded to 32); token = n
  v16h bqt = __builtin_shufflevector(
      *(const v8h*)(y1h + (rowBase + n) * 64 + hh * 16 + mb), zero8h(),
      0,1,2,3,4,5,6,7,8,9,10,11,12,13,14,15);
  // secm^T A-fragment (K zero-padded): a[h<8] = secmT[m][kb+h]
  v16h asec = load_A_pad16(secmh_t + (size_t)(b * kHeads + hh) * 256, 0, 0, 16);

  v8f acc = zero8();                 // out1^T: rows = feature d, cols = token
  float mcur = -1e30f, lcur = 0.f;   // per-lane stats for token i = n

#pragma unroll 1
  for (int jt = 0; jt < kN / 32; ++jt) {
    size_t jbase = (size_t)b * kN + jt * 32;
    if (jt + 1 < kN / 32) {          // stream-ahead into cache
      __builtin_prefetch(y2h + (jbase + 32) * 64 + lane * 4, 0, 1);
      __builtin_prefetch(x1swA + ((size_t)((b * 72 + jt + 1) * 4 + hh) * 32 + lane) * 16, 0, 1);
    }
    // K_j A-fragments (two 16-row tiles, K zero-padded to 32); y2 pre-scaled
    v16h ak0 = load_A_pad16(y2h, (int)jbase,      hh * 16, 64);
    v16h ak1 = load_A_pad16(y2h, (int)jbase + 16, hh * 16, 64);
    v8f s0 = wmma_f16(ak0, bqt, zero8());   // rows j = 0..15,  cols = token
    v8f s1 = wmma_f16(ak1, bqt, zero8());   // rows j = 16..31, cols = token

    // base-2 online softmax for token n: 16 local values + lane-pair combine
    float mloc = -1e30f;
#pragma unroll
    for (int r = 0; r < 8; ++r) mloc = fmaxf(mloc, fmaxf(s0[r], s1[r]));
    mloc = fmaxf(mloc, __shfl_xor(mloc, 16, 32));
    float mnew = fmaxf(mcur, mloc);
    float corr = fast_exp2(mcur - mnew);
    float lsum = 0.f;
    v8h p0h, p1h;
#pragma unroll
    for (int r = 0; r < 8; ++r) {
      float p0 = fast_exp2(s0[r] - mnew);
      float p1 = fast_exp2(s1[r] - mnew);
      lsum += p0 + p1;
      p0h[r] = (_Float16)p0;
      p1h[r] = (_Float16)p1;
    }
    lsum += __shfl_xor(lsum, 16, 32);
    lcur = lcur * corr + lsum;
    mcur = mnew;
#pragma unroll
    for (int r = 0; r < 8; ++r) acc[r] *= corr;

    // stage P row-major [i=n][j]: two packed 16B LDS stores per lane.
    *(v8h*)(&Pl[w][n * 32 + mb])      = p0h;
    *(v8h*)(&Pl[w][n * 32 + 16 + mb]) = p1h;
    // Wave-local ordering: DS ops are in-order per wave; drain DScnt and fence
    // the compiler so the cross-lane reload isn't hoisted above the stores.
    asm volatile("s_wait_dscnt 0x0" ::: "memory");
    // B-fragment of P^T: b[h<8] = P[i=n][j=mb+h], b[h>=8] = P[n][16+mb+h-8]
    v16h bp = __builtin_shufflevector(*(const v8h*)(&Pl[w][n * 32 + mb]),
                                      *(const v8h*)(&Pl[w][n * 32 + 16 + mb]),
                                      0,1,2,3,4,5,6,7,8,9,10,11,12,13,14,15);
    asm volatile("" ::: "memory");   // WAR fence vs next iteration's stores
    // A-fragment of X1^T (pre-swizzled, contiguous)
    v16h ax = *(const v16h*)(x1swA + ((size_t)((b * 72 + jt) * 4 + hh) * 32 + lane) * 16);
    acc = wmma_f16(ax, bp, acc);
  }

  // out2^T = secm^T @ Q^T ; combine and store packed (8 features per lane)
  v8f o2 = wmma_f16(asec, bqt, zero8());
  v8h outv;
  float linv = 1.f / lcur;
#pragma unroll
  for (int r = 0; r < 8; ++r)
    outv[r] = (_Float16)((acc[r] * linv) * o2[r]);
  *(v8h*)(attnh + (rowBase + n) * 64 + hh * 16 + mb) = outv;
}

// ---------------- K4: out-proj + residual + LN2 + MLP + residual + transpose ----
__global__ void mlp_kernel(const _Float16* __restrict__ attnh,
                           const float* __restrict__ xt,
                           const _Float16* __restrict__ wouth,
                           const float* __restrict__ b_out,
                           const float* __restrict__ g2, const float* __restrict__ bb2,
                           const _Float16* __restrict__ w1h, const float* __restrict__ b1,
                           const _Float16* __restrict__ w2h, const float* __restrict__ b2,
                           float* __restrict__ out) {
  __shared__ float    lds_x[4][16 * 64];
  __shared__ _Float16 lds_h[4][16 * 64];
  __shared__ _Float16 lds_H[4][16 * 256];
  int w = threadIdx.x >> 5;
  int rowTile = blockIdx.x * 4 + w;
  int row0 = rowTile * 16;
  int lane = threadIdx.x & 31, n = lane & 15, mb = (lane >> 4) << 3;

  // Stage 1: xt2 = xt + attn @ w_out + b_out  -> LDS (fp32)
  {
    v16h a0 = load_A(attnh, row0, 0, 64);
    v16h a1 = load_A(attnh, row0, 32, 64);
#pragma unroll
    for (int nt = 0; nt < 4; ++nt) {
      v8f c = wmma_f16(a0, load_B_sw(wouth, 0, nt, 4), zero8());
      c     = wmma_f16(a1, load_B_sw(wouth, 1, nt, 4), c);
      float bo = b_out[nt * 16 + n];
#pragma unroll
      for (int r = 0; r < 8; ++r) {
        int M = mb + r;
        lds_x[w][M * 64 + nt * 16 + n] =
            xt[(size_t)(row0 + M) * 64 + nt * 16 + n] + c[r] + bo;
      }
    }
  }
  __syncthreads();

  // Stage 2: LayerNorm2 per row -> f16 in LDS
  if (lane < 16) {
    int M = lane;
    float s = 0.f, s2 = 0.f;
#pragma unroll 1
    for (int cc = 0; cc < 64; ++cc) { float v = lds_x[w][M * 64 + cc]; s += v; s2 += v * v; }
    float m  = s * (1.f/64.f);
    float vv = s2 * (1.f/64.f) - m * m;
    float rs = rsqrtf(vv + 1e-5f);
#pragma unroll 1
    for (int cc = 0; cc < 64; ++cc) {
      float v = lds_x[w][M * 64 + cc];
      lds_h[w][M * 64 + cc] = (_Float16)((v - m) * rs * g2[cc] + bb2[cc]);
    }
  }
  __syncthreads();

  // Stage 3: H = leaky_relu(hin @ w1 + b1)  (16 x 256)
  {
    v16h a0 = load_A(&lds_h[w][0], 0, 0, 64);
    v16h a1 = load_A(&lds_h[w][0], 0, 32, 64);
#pragma unroll 1
    for (int nt = 0; nt < 16; ++nt) {
      v8f c = wmma_f16(a0, load_B_sw(w1h, 0, nt, 16), zero8());
      c     = wmma_f16(a1, load_B_sw(w1h, 1, nt, 16), c);
      float b1v = b1[nt * 16 + n];
#pragma unroll
      for (int r = 0; r < 8; ++r) {
        float v = c[r] + b1v;
        v = (v > 0.f) ? v : 0.01f * v;
        lds_H[w][(mb + r) * kMLP + nt * 16 + n] = (_Float16)v;
      }
    }
  }
  __syncthreads();

  // Stage 4: final = xt2 + H @ w2 + b2, store transposed to (B,C,H,W)
#pragma unroll
  for (int nt = 0; nt < 4; ++nt) {
    v8f c = zero8();
#pragma unroll
    for (int kt = 0; kt < 8; ++kt)
      c = wmma_f16(load_A(&lds_H[w][0], 0, kt * 32, kMLP),
                   load_B_sw(w2h, kt, nt, 4), c);
    float b2v = b2[nt * 16 + n];
#pragma unroll
    for (int r = 0; r < 8; ++r) {
      int M = mb + r;
      float v = lds_x[w][M * 64 + nt * 16 + n] + c[r] + b2v;
      int g = row0 + M;
      int bidx = g / kN, p = g % kN;
      out[(size_t)bidx * kC * kN + (size_t)(nt * 16 + n) * kN + p] = v;
    }
  }
}

// ---------------- host launcher ----------------
extern "C" void kernel_launch(void* const* d_in, const int* in_sizes, int n_in,
                              void* d_out, int out_size, void* d_ws, size_t ws_size,
                              hipStream_t stream) {
  (void)in_sizes; (void)n_in; (void)out_size; (void)ws_size;
  const float* x     = (const float*)d_in[0];
  const float* y     = (const float*)d_in[1];
  const float* ln1_g = (const float*)d_in[2];
  const float* ln1_b = (const float*)d_in[3];
  const float* w_sa1 = (const float*)d_in[4];
  const float* w_sa2 = (const float*)d_in[5];
  const float* w_se1 = (const float*)d_in[6];
  const float* w_se2 = (const float*)d_in[7];
  const float* w_out = (const float*)d_in[8];
  const float* b_out = (const float*)d_in[9];
  const float* ln2_g = (const float*)d_in[10];
  const float* ln2_b = (const float*)d_in[11];
  const float* w1    = (const float*)d_in[12];
  const float* b1    = (const float*)d_in[13];
  const float* w2    = (const float*)d_in[14];
  const float* b2    = (const float*)d_in[15];
  float* out = (float*)d_out;

  char* ws = (char*)d_ws;
  size_t off = 0;
  auto carve = [&](size_t bytes) -> char* {
    char* p = ws + off;
    off = (off + bytes + 255) & ~(size_t)255;
    return p;
  };
  float*    xt     = (float*)   carve((size_t)kRows * 64 * 4);
  _Float16* xnh    = (_Float16*)carve((size_t)kRows * 64 * 2);
  _Float16* ynh    = (_Float16*)carve((size_t)kRows * 64 * 2);
  _Float16* y1h    = (_Float16*)carve((size_t)kRows * 64 * 2);
  _Float16* y2h    = (_Float16*)carve((size_t)kRows * 64 * 2);
  _Float16* x1h    = (_Float16*)carve((size_t)kRows * 64 * 2);
  _Float16* x2h    = (_Float16*)carve((size_t)kRows * 64 * 2);
  _Float16* x1swA  = (_Float16*)carve((size_t)kRows * 64 * 2);
  _Float16* attnh  = (_Float16*)carve((size_t)kRows * 64 * 2);
  _Float16* secmt  = (_Float16*)carve((size_t)kB * kHeads * 256 * 2);
  _Float16* wa1h   = (_Float16*)carve(4096 * 2);
  _Float16* wa2h   = (_Float16*)carve(4096 * 2);
  _Float16* we1h   = (_Float16*)carve(4096 * 2);
  _Float16* we2h   = (_Float16*)carve(4096 * 2);
  _Float16* wouth  = (_Float16*)carve(4096 * 2);
  _Float16* w1h    = (_Float16*)carve(16384 * 2);
  _Float16* w2h    = (_Float16*)carve(16384 * 2);

  // weights -> f16, pre-swizzled to B-fragment layout
  cvt_swz<<<16, 256, 0, stream>>>(w_sa1, wa1h, 64, 4, 4096);
  cvt_swz<<<16, 256, 0, stream>>>(w_sa2, wa2h, 64, 4, 4096);
  cvt_swz<<<16, 256, 0, stream>>>(w_se1, we1h, 64, 4, 4096);
  cvt_swz<<<16, 256, 0, stream>>>(w_se2, we2h, 64, 4, 4096);
  cvt_swz<<<16, 256, 0, stream>>>(w_out, wouth, 64, 4, 4096);
  cvt_swz<<<64, 256, 0, stream>>>(w1, w1h, 256, 16, 16384);
  cvt_swz<<<64, 256, 0, stream>>>(w2, w2h, 64, 4, 16384);

  ln_kernel<<<(kRows + 255) / 256, 256, 0, stream>>>(x, y, ln1_g, ln1_b, xt, xnh, ynh);
  proj_kernel<<<144, 256, 0, stream>>>(xnh, ynh, wa1h, wa2h, we1h, we2h,
                                       y1h, y2h, x1h, x2h);
  swizzle_x1A<<<(kRows * 64) / 256, 256, 0, stream>>>(x1h, x1swA);
  secm_kernel<<<1, 256, 0, stream>>>(x1h, x2h, secmt);
  attn_kernel<<<288, 128, 0, stream>>>(y1h, y2h, x1swA, secmt, attnh);
  mlp_kernel<<<72, 128, 0, stream>>>(attnh, xt, wouth, b_out, ln2_g, ln2_b,
                                     w1h, b1, w2h, b2, out);
}
